// ELAB_59133109731761
// MI455X (gfx1250) — matrix operand order repaired and checked by reading, not compiled, not run on hardware
//
#include <hip/hip_runtime.h>
#include <hip/hip_bf16.h>
#include <stdint.h>

// ---------------- CDNA5 fragment types ----------------
typedef __attribute__((ext_vector_type(16))) __bf16 v16bf;
typedef __attribute__((ext_vector_type(8)))  float  v8f;

#define HDIM 256
#define WDIM 256
#define HW   (HDIM*WDIM)
#define BATCH 2

__device__ __forceinline__ uint32_t f2bf1(float f){
  uint32_t u = __float_as_uint(f);
  uint32_t r = u + 0x7FFFu + ((u >> 16) & 1u);   // round-to-nearest-even
  return r >> 16;
}
__device__ __forceinline__ uint32_t packbf(float lo, float hi){
  return (f2bf1(hi) << 16) | (f2bf1(lo) & 0xFFFFu);
}

union FragBF { uint32_t u[8]; v16bf v; };

// Branchless shifted gather: load address is ALWAYS valid (clamped);
// OOB / padded-K elements are zeroed via select -> no saveexec blocks.
template<int KIN, int SHIFT_G>
__device__ __forceinline__ float gatherX(const float* __restrict__ X,
                                         int b, int c, int h, int w){
  int dw = 0, dh = 0;
  if (SHIFT_G > 0){
    dw = (c < SHIFT_G) ? 1 : ((c < 2*SHIFT_G) ? -1 : 0);
    dh = (c >= 2*SHIFT_G && c < 3*SHIFT_G) ? 1 :
         ((c >= 3*SHIFT_G && c < 4*SHIFT_G) ? -1 : 0);
  }
  int ww = w + dw, hh = h + dh;
  bool oob = (c >= KIN) | (ww < 0) | (ww >= WDIM) | (hh < 0) | (hh >= HDIM);
  int cc = oob ? 0 : c;
  int wc = oob ? w : ww;
  int hc = oob ? h : hh;
  float v = X[((size_t)b*KIN + cc)*HW + (size_t)hc*WDIM + wc];
  return oob ? 0.f : v;
}

// ---------------- fused 1x1-conv GEMM via WMMA bf16 ----------------
// Block = 4 waves = PPB pixel-tiles x MGROUPS m-groups. The 32x16 activation
// slab is staged ONCE per block into LDS (packed bf16 pairs, double buffered),
// then each wave feeds MT WMMAs per K-step from 8 ds_load_b32.
template<int KIN, int KPAD, int MOUT, int MGROUPS, int MT,
         int SHIFT_G, bool RELU, bool RES>
__global__ __launch_bounds__(128) void gemm1x1_wmma(
    const float*    __restrict__ X,
    const uint32_t* __restrict__ Wp,     // bf16 pairs, [Mpad][KPAD], zero padded
    const float*    __restrict__ bias,
    const float*    __restrict__ res,
    float*          __restrict__ out)
{
  constexpr int PPB = 4 / MGROUPS;       // pixel tiles per block
  constexpr int PRS = 19;                // padded pair-row stride (dwords)
  __shared__ uint32_t slab[2][PPB][16*PRS];

  const int TPR = HDIM*(WDIM/16);        // 4096 pixel tiles per batch image
  int tid  = threadIdx.x;
  int wv   = tid >> 5;
  int lane = tid & 31;
  int nloc = lane & 15;
  int hi   = lane >> 4;
  const int koff = hi ? 8 : 0;

  // block's pixel-tile coordinates
  int pb[PPB], ph[PPB], pw[PPB];
  #pragma unroll
  for (int pt = 0; pt < PPB; pt++){
    int ptile = blockIdx.x*PPB + pt;
    pb[pt] = ptile / TPR;
    int r  = ptile % TPR;
    ph[pt] = r / (WDIM/16);
    pw[pt] = (r % (WDIM/16)) * 16;
  }

  // cooperative stage of one 32ch x 16pix slab (packed bf16 pairs)
  auto stage = [&](int bufi, int kk){
    #pragma unroll
    for (int it = 0; it < 2*PPB; it++){
      int e   = tid + it*128;            // e < PPB*256
      int pt  = e >> 8;
      int rem = e & 255;
      int pr  = rem >> 4;                // pair row 0..15
      int n   = rem & 15;
      int c0  = kk + pr*2;
      float lo  = gatherX<KIN,SHIFT_G>(X, pb[pt], c0,     ph[pt], pw[pt]+n);
      float hi2 = gatherX<KIN,SHIFT_G>(X, pb[pt], c0 + 1, ph[pt], pw[pt]+n);
      slab[bufi][pt][pr*PRS + n] = packbf(lo, hi2);
    }
  };

  // this wave's assignment
  int ptl = wv / MGROUPS;
  int mg  = wv % MGROUPS;
  int b = pb[ptl], h = ph[ptl], w = pw[ptl] + nloc;
  const int mbase = mg * MT * 16;

  v8f acc[MT];
  #pragma unroll
  for (int t = 0; t < MT; t++) acc[t] = (v8f){};

  const uint32_t* wrow[MT];
  #pragma unroll
  for (int t = 0; t < MT; t++)
    wrow[t] = Wp + (((size_t)(mbase + t*16 + nloc) * KPAD) >> 1);

  stage(0, 0);
  __syncthreads();

  int buf = 0;
  for (int k0 = 0; k0 < KPAD; k0 += 32, buf ^= 1){
    if (k0 + 32 < KPAD) stage(buf ^ 1, k0 + 32);     // overlap next slab
    if (k0 + 64 < KIN)                               // CDNA5 global_prefetch_b8
      __builtin_prefetch(&X[((size_t)b*KIN + (k0+64))*HW + (size_t)h*WDIM + w], 0, 1);

    // B 32x16 bf16 fragment from LDS: pair row = hi*4 + j + (j&4), col = nloc
    FragBF Bf;
    #pragma unroll
    for (int j = 0; j < 8; j++)
      Bf.u[j] = slab[buf][ptl][(hi*4 + j + (j & 4))*PRS + nloc];

    // A 16x32 bf16: lanes 0-15 K 0..7/16..23 ; lanes 16-31 K 8..15/24..31
    #pragma unroll
    for (int t = 0; t < MT; t++){
      FragBF A;
      #pragma unroll
      for (int j = 0; j < 4; j++) A.u[j]   = wrow[t][(k0 +      koff + 2*j) >> 1];
      #pragma unroll
      for (int j = 0; j < 4; j++) A.u[4+j] = wrow[t][(k0 + 16 + koff + 2*j) >> 1];
      acc[t] = __builtin_amdgcn_wmma_f32_16x16x32_bf16(
                   false, A.v, false, Bf.v, (short)0, acc[t], false, false);
    }
    __syncthreads();   // staging of next buffer + all reads of current done
  }

  // C/D layout: VGPR rr -> M = mbase + 16t + rr + 8*hi ; N = lane&15
  #pragma unroll
  for (int t = 0; t < MT; t++){
    #pragma unroll
    for (int rr = 0; rr < 8; rr++){
      int m = mbase + t*16 + rr + hi*8;
      if (m < MOUT){
        size_t oidx = ((size_t)b*MOUT + m)*HW + (size_t)h*WDIM + w;
        float v = acc[t][rr] + bias[m];
        if (RES)  v += res[oidx];
        if (RELU) v  = fmaxf(v, 0.f);
        out[oidx] = v;
      }
    }
  }
}

// ---------------- weight fp32 -> padded bf16 ----------------
__global__ void convert_w(const float* __restrict__ w, uint16_t* __restrict__ wbf,
                          int M, int K, int Mpad, int Kpad){
  int total = Mpad * Kpad;
  for (int i = blockIdx.x*blockDim.x + threadIdx.x; i < total; i += gridDim.x*blockDim.x){
    int m = i / Kpad, k = i % Kpad;
    float v = (m < M && k < K) ? w[(size_t)m*K + k] : 0.f;
    wbf[i] = (uint16_t)f2bf1(v);
  }
}

// ---------------- BatchNorm statistics ----------------
__global__ __launch_bounds__(256) void bn_stats(const float* __restrict__ z,
                                                float* __restrict__ mean,
                                                float* __restrict__ invstd){
  int c = blockIdx.x;                    // 0..359
  float s = 0.f, s2 = 0.f;
  for (int b = 0; b < BATCH; b++){
    const float* p = z + ((size_t)b*360 + c)*HW;
    for (int i = threadIdx.x; i < HW; i += 256){ float v = p[i]; s += v; s2 = fmaf(v, v, s2); }
  }
  __shared__ float sh[256], sh2[256];
  sh[threadIdx.x] = s; sh2[threadIdx.x] = s2; __syncthreads();
  for (int st = 128; st > 0; st >>= 1){
    if (threadIdx.x < st){ sh[threadIdx.x] += sh[threadIdx.x+st]; sh2[threadIdx.x] += sh2[threadIdx.x+st]; }
    __syncthreads();
  }
  if (threadIdx.x == 0){
    float inv_n = 1.f / (float)((size_t)BATCH*HW);
    float mu  = sh[0] * inv_n;
    float var = sh2[0] * inv_n - mu*mu;
    mean[c] = mu; invstd[c] = rsqrtf(var + 1e-5f);
  }
}

// ---------------- CDNA5 async global->LDS helpers ----------------
__device__ __forceinline__ void async_g2l_b32(const float* gsrc, const float* ldst){
  uint32_t loff = (uint32_t)(uintptr_t)ldst;   // flat LDS aperture: addr[31:0] = LDS offset
  asm volatile("global_load_async_to_lds_b32 %0, %1, off"
               :: "v"(loff), "v"(gsrc) : "memory");
}
__device__ __forceinline__ void wait_asynccnt0(){
  asm volatile("s_wait_asynccnt 0x0" ::: "memory");
}

// ---------------- windowed self-attention (BN fused at load) ----------------
__global__ __launch_bounds__(64) void win_attn(const float* __restrict__ z,
                                               float* __restrict__ out,
                                               const float* __restrict__ mean,
                                               const float* __restrict__ invstd,
                                               const float* __restrict__ g,
                                               const float* __restrict__ bb,
                                               int chunk, int wsz){
  int s    = wsz >> 1;
  int nw   = HDIM / wsz;
  int npix = wsz * wsz;
  int wperb= 64 / npix;
  int lw   = threadIdx.x / npix;
  int n    = threadIdx.x % npix;
  int wid  = blockIdx.x * wperb + lw;          // global window id
  int b    = wid / (nw*nw);
  int r2   = wid % (nw*nw);
  int wh   = r2 / nw, wwc = r2 % nw;
  int rr = n / wsz, cc = n % wsz;
  int h = (wh*wsz + rr + s) & (HDIM-1);
  int w = (wwc*wsz + cc + s) & (WDIM-1);
  int base = lw * npix;
  size_t pix = (size_t)h*WDIM + w;

  __shared__ float qs[64][10];
  __shared__ float vs[64][10];

  for (int hd = 0; hd < 6; hd++){
    // async stage raw z -> LDS (ASYNCcnt path)
    #pragma unroll
    for (int c = 0; c < 10; c++){
      int qc = chunk*120      + hd*10 + c;
      int vc = chunk*120 + 60 + hd*10 + c;
      async_g2l_b32(&z[((size_t)b*360 + qc)*HW + pix], &qs[base+n][c]);
      async_g2l_b32(&z[((size_t)b*360 + vc)*HW + pix], &vs[base+n][c]);
    }
    wait_asynccnt0();
    __syncthreads();
    // fused BatchNorm normalization in place
    #pragma unroll
    for (int c = 0; c < 10; c++){
      int qc = chunk*120      + hd*10 + c;
      int vc = chunk*120 + 60 + hd*10 + c;
      qs[base+n][c] = (qs[base+n][c] - mean[qc]) * invstd[qc] * g[qc] + bb[qc];
      vs[base+n][c] = (vs[base+n][c] - mean[vc]) * invstd[vc] * g[vc] + bb[vc];
    }
    __syncthreads();

    float sc[64]; float mx = -3.4e38f;
    for (int m = 0; m < npix; m++){
      float d = 0.f;
      #pragma unroll
      for (int c = 0; c < 10; c++) d += qs[base+n][c]*qs[base+m][c];
      sc[m] = d; mx = fmaxf(mx, d);
    }
    float Zs = 0.f;
    for (int m = 0; m < npix; m++){ float e = __expf(sc[m]-mx); sc[m] = e; Zs += e; }
    float inv = 1.f / Zs;
    #pragma unroll
    for (int c = 0; c < 10; c++){
      float o = 0.f;
      for (int m = 0; m < npix; m++) o += sc[m]*vs[base+m][c];
      out[((size_t)b*180 + chunk*60 + hd*10 + c)*HW + pix] = o*inv;
    }
    __syncthreads();
  }
}

// ---------------- host launcher ----------------
extern "C" void kernel_launch(void* const* d_in, const int* in_sizes, int n_in,
                              void* d_out, int out_size, void* d_ws, size_t ws_size,
                              hipStream_t stream) {
  (void)in_sizes; (void)n_in; (void)out_size; (void)ws_size;
  const float* x    = (const float*)d_in[0];
  const float* w0   = (const float*)d_in[1];
  const float* b0   = (const float*)d_in[2];
  const float* w1   = (const float*)d_in[3];
  const float* b1   = (const float*)d_in[4];
  const float* w_in = (const float*)d_in[5];
  const float* b_in = (const float*)d_in[6];
  const float* bn_g = (const float*)d_in[7];
  const float* bn_b = (const float*)d_in[8];
  const float* w_out= (const float*)d_in[9];
  const float* b_out= (const float*)d_in[10];

  // workspace layout (z aliases y1: y1 fully consumed before z is written)
  float*    y1z   = (float*)d_ws;                        // [2][360][HW]
  float*    x2    = y1z + (size_t)BATCH*360*HW;          // [2][180][HW]
  float*    att   = x2  + (size_t)BATCH*180*HW;          // [2][180][HW]
  uint16_t* w0bf  = (uint16_t*)(att + (size_t)BATCH*180*HW); // 384x192
  uint16_t* w1bf  = w0bf  + 384*192;                     // 192x384
  uint16_t* winbf = w1bf  + 192*384;                     // 384x192
  uint16_t* woutbf= winbf + 384*192;                     // 192x192
  float*    mean  = (float*)(woutbf + 192*192);
  float*    invstd= mean + 360;

  convert_w<<<112, 256, 0, stream>>>(w0,    w0bf,   360, 180, 384, 192);
  convert_w<<<112, 256, 0, stream>>>(w1,    w1bf,   180, 360, 192, 384);
  convert_w<<<112, 256, 0, stream>>>(w_in,  winbf,  360, 180, 384, 192);
  convert_w<<< 64, 256, 0, stream>>>(w_out, woutbf, 180, 180, 192, 192);

  const int PT = BATCH*HDIM*(WDIM/16);   // 8192 pixel tiles

  // y1 = relu(w0 @ shift(x) + b0)   K=180 G=36, M=360: 4 mgroups, 1 ptile/block
  gemm1x1_wmma<180,192,360,4,6,36,true,false>
      <<<PT, 128, 0, stream>>>(x, (const uint32_t*)w0bf, b0, nullptr, y1z);

  // x2 = w1 @ shift(y1) + b1 + x    K=360 G=72, M=180: 2 mgroups, 2 ptiles/block
  gemm1x1_wmma<360,384,180,2,6,72,false,true>
      <<<PT/2, 128, 0, stream>>>(y1z, (const uint32_t*)w1bf, b1, x, x2);

  // z = w_in @ x2 + b_in            (z aliases y1z)
  gemm1x1_wmma<180,192,360,4,6,0,false,false>
      <<<PT, 128, 0, stream>>>(x2, (const uint32_t*)winbf, b_in, nullptr, y1z);

  bn_stats<<<360, 256, 0, stream>>>(y1z, mean, invstd);

  // window attention (BN fused): chunks 0/1/2, ws=2/4/8 -> 2048 blocks of 64
  win_attn<<<2048, 64, 0, stream>>>(y1z, att, mean, invstd, bn_g, bn_b, 0, 2);
  win_attn<<<2048, 64, 0, stream>>>(y1z, att, mean, invstd, bn_g, bn_b, 1, 4);
  win_attn<<<2048, 64, 0, stream>>>(y1z, att, mean, invstd, bn_g, bn_b, 2, 8);

  // out = w_out @ att + b_out + x2
  gemm1x1_wmma<180,192,180,2,6,0,false,true>
      <<<PT/2, 128, 0, stream>>>(att, (const uint32_t*)woutbf, b_out, x2, (float*)d_out);
}
